// Cross_Path_6193342841396
// MI455X (gfx1250) — matrix operand order repaired and checked by reading, not compiled, not run on hardware
//
#include <hip/hip_runtime.h>
#include <hip/hip_bf16.h>

typedef __attribute__((ext_vector_type(16))) _Float16 v16h;
typedef __attribute__((ext_vector_type(8)))  float    v8f;

// ---------------------------------------------------------------------------
// Generic WMMA GEMM:  C[M,N] = act( A[M,K] * B + bias )
//  - A: row-major tokens (lda) or channel-major [b][K][HW] when chHW>0
//  - B: bWeight=1 -> Bm is [N,K] row-major (torch Linear weight, b[k][n]=Bm[n][k])
//       bWeight=0 -> Bm is [K,N] row-major
//  - batched over blockIdx.z with element strides sAz/sBz/sCz
// Block: 256 threads = 8 waves; each wave computes a 16(M) x 64(N) strip.
// Requires M%128==0, N%64==0, K%32==0 (true for every call site here).
// ---------------------------------------------------------------------------
__global__ __launch_bounds__(256) void gemm_wmma_kernel(
    const float* __restrict__ A, long lda, long sAz, int chHW,
    const float* __restrict__ Bm, int ldb, long sBz, int bWeight,
    const float* __restrict__ bias,
    float* __restrict__ C, long ldc, long sCz,
    int M, int N, int K, int act)
{
  __shared__ _Float16 As[128][40];
  __shared__ _Float16 Bs[64][40];
  const int tid    = threadIdx.x;
  const int wave   = tid >> 5;
  const int lane   = tid & 31;
  const int lane15 = lane & 15;
  const int laneHi = lane >> 4;
  const int m0 = blockIdx.x * 128;
  const int n0 = blockIdx.y * 64;
  const float* Ab = A  + (long)blockIdx.z * sAz;
  const float* Bb = Bm + (long)blockIdx.z * sBz;
  float*       Cb = C  + (long)blockIdx.z * sCz;

  v8f acc[4] = {};

  const int mlA = tid >> 1;
  const int kbA = (tid & 1) * 16;

  for (int kt = 0; kt < K; kt += 32) {
    // ---- stage A tile (128 x 32) as f16 ----
    {
      const int mg = m0 + mlA;
      if (chHW > 0) {
        const int bI  = mg / chHW;
        const int pix = mg - bI * chHW;
        const float* ap = Ab + (long)bI * (long)K * (long)chHW + pix;
#pragma unroll
        for (int j = 0; j < 16; ++j)
          As[mlA][kbA + j] = (_Float16)ap[(long)(kt + kbA + j) * (long)chHW];
      } else {
        const float* ap = Ab + (long)mg * lda + kt + kbA;
        __builtin_prefetch((const void*)(ap + 32), 0, 1);   // gfx1250 global_prefetch_b8
#pragma unroll
        for (int j = 0; j < 16; ++j)
          As[mlA][kbA + j] = (_Float16)ap[j];
      }
    }
    // ---- stage B tile (32k x 64n), stored n-major Bs[n][k] ----
    if (bWeight) {
      const int nl = tid >> 2, kb = (tid & 3) * 8;
      const float* bp = Bb + (long)(n0 + nl) * ldb + kt + kb;
#pragma unroll
      for (int j = 0; j < 8; ++j) Bs[nl][kb + j] = (_Float16)bp[j];
    } else {
      const int kl = tid >> 3, nb = (tid & 7) * 8;
      const float* bp = Bb + (long)(kt + kl) * ldb + n0 + nb;
#pragma unroll
      for (int j = 0; j < 8; ++j) Bs[nb + j][kl] = (_Float16)bp[j];
    }
    __syncthreads();

    // A fragment (contiguous k, matching B):
    // lane<16: k {0..15}; lane>=16: k {16..31}
    const int row = wave * 16 + lane15;
    const int ak0 = laneHi * 16;
    v16h a;
#pragma unroll
    for (int j = 0; j < 8; ++j) { a[j] = As[row][ak0 + j]; a[8 + j] = As[row][ak0 + 8 + j]; }

    // B fragment: b[i] = B[k = i + 16*laneHi][n = lane15]  (contiguous in Bs[n][k])
    const int bk0 = laneHi * 16;
#pragma unroll
    for (int s = 0; s < 4; ++s) {
      const int col = s * 16 + lane15;
      v16h b;
#pragma unroll
      for (int j = 0; j < 16; ++j) b[j] = Bs[col][bk0 + j];
      acc[s] = __builtin_amdgcn_wmma_f32_16x16x32_f16(false, a, false, b,
                                                      (short)0, acc[s], false, false);
    }
    __syncthreads();
  }

  // C/D layout: VGPR v, lane<16 -> M=v, lane>=16 -> M=v+8; N = lane15
#pragma unroll
  for (int s = 0; s < 4; ++s) {
    const int col = n0 + s * 16 + lane15;
    const float bv = bias ? bias[col] : 0.0f;
#pragma unroll
    for (int v = 0; v < 8; ++v) {
      const int rowg = m0 + wave * 16 + v + 8 * laneHi;
      float x = acc[s][v] + bv;
      if (act == 1) x = fmaxf(x, 0.0f);
      Cb[(long)rowg * ldc + col] = x;
    }
  }
}

// ---------------------------------------------------------------------------
// Implicit-GEMM 3x3 SAME conv + BN(eval) + ReLU6, NCHW in/out, via WMMA.
// M = B*H*W output pixels, N = Cout, K = Cin*9.
// ---------------------------------------------------------------------------
__global__ __launch_bounds__(256) void conv3x3_wmma_kernel(
    const float* __restrict__ In, const float* __restrict__ Wt,
    const float* __restrict__ g, const float* __restrict__ beta,
    const float* __restrict__ mean, const float* __restrict__ var,
    float* __restrict__ Out, int Cin, int Cout, int H, int Wd)
{
  __shared__ _Float16 As[128][40];
  __shared__ _Float16 Bs[64][40];
  const int tid    = threadIdx.x;
  const int wave   = tid >> 5;
  const int lane   = tid & 31;
  const int lane15 = lane & 15;
  const int laneHi = lane >> 4;
  const int m0 = blockIdx.x * 128;
  const int n0 = blockIdx.y * 64;
  const int HW = H * Wd;
  const int K  = Cin * 9;

  v8f acc[4] = {};

  const int mlA = tid >> 1;
  const int kbA = (tid & 1) * 16;
  const int mg  = m0 + mlA;
  const int bI  = mg / HW;
  const int pix = mg - bI * HW;
  const int py  = pix / Wd;
  const int px  = pix - py * Wd;

  for (int kt = 0; kt < K; kt += 32) {
    // ---- gather A tile (im2col on the fly) ----
#pragma unroll
    for (int j = 0; j < 16; ++j) {
      const int k  = kt + kbA + j;
      const int c  = k / 9;
      const int r  = k - c * 9;
      const int dy = r / 3 - 1;
      const int dx = r - (r / 3) * 3 - 1;
      const int yy = py + dy, xx = px + dx;
      float v = 0.0f;
      if (yy >= 0 && yy < H && xx >= 0 && xx < Wd)
        v = In[(((long)bI * Cin + c) * H + yy) * Wd + xx];
      As[mlA][kbA + j] = (_Float16)v;
    }
    // ---- stage weights (Wt is [Cout][Cin*9] flat) ----
    {
      const int nl = tid >> 2, kb = (tid & 3) * 8;
      const float* bp = Wt + (long)(n0 + nl) * K + kt + kb;
#pragma unroll
      for (int j = 0; j < 8; ++j) Bs[nl][kb + j] = (_Float16)bp[j];
    }
    __syncthreads();

    const int row = wave * 16 + lane15;
    const int ak0 = laneHi * 16;
    v16h a;
#pragma unroll
    for (int j = 0; j < 8; ++j) { a[j] = As[row][ak0 + j]; a[8 + j] = As[row][ak0 + 8 + j]; }
    const int bk0 = laneHi * 16;
#pragma unroll
    for (int s = 0; s < 4; ++s) {
      const int col = s * 16 + lane15;
      v16h b;
#pragma unroll
      for (int j = 0; j < 16; ++j) b[j] = Bs[col][bk0 + j];
      acc[s] = __builtin_amdgcn_wmma_f32_16x16x32_f16(false, a, false, b,
                                                      (short)0, acc[s], false, false);
    }
    __syncthreads();
  }

#pragma unroll
  for (int s = 0; s < 4; ++s) {
    const int col = n0 + s * 16 + lane15;
    const float sc = g[col] * rsqrtf(var[col] + 1e-5f);
    const float mu = mean[col];
    const float bb = beta[col];
#pragma unroll
    for (int v = 0; v < 8; ++v) {
      const int rowg = m0 + wave * 16 + v + 8 * laneHi;
      const int ob   = rowg / HW;
      const int opix = rowg - ob * HW;
      float x = (acc[s][v] - mu) * sc + bb;
      x = fminf(fmaxf(x, 0.0f), 6.0f);                    // ReLU6
      Out[((long)ob * Cout + col) * HW + opix] = x;
    }
  }
}

// ---------------------------------------------------------------------------
// Bilinear 2x upsample, align_corners=True, NCHW.
// ---------------------------------------------------------------------------
__global__ void up2_kernel(const float* __restrict__ in, float* __restrict__ out,
                           int B, int C, int H, int W)
{
  const long total = (long)B * C * 4 * H * W;
  long i = (long)blockIdx.x * blockDim.x + threadIdx.x;
  if (i >= total) return;
  const int OW = 2 * W, OH = 2 * H;
  const int ox = (int)(i % OW); long t = i / OW;
  const int oy = (int)(t % OH); t /= OH;
  const int c  = (int)(t % C);
  const int b  = (int)(t / C);
  const float fy = oy * (float)(H - 1) / (float)(OH - 1);
  const float fx = ox * (float)(W - 1) / (float)(OW - 1);
  int ly = (int)floorf(fy), lx = (int)floorf(fx);
  const float wy = fy - ly, wx = fx - lx;
  int hy = ly + 1; if (hy > H - 1) hy = H - 1;
  int hx = lx + 1; if (hx > W - 1) hx = W - 1;
  const float* ip = in + ((long)b * C + c) * H * W;
  const float v00 = ip[ly * W + lx], v01 = ip[ly * W + hx];
  const float v10 = ip[hy * W + lx], v11 = ip[hy * W + hx];
  out[i] = (1.0f - wy) * ((1.0f - wx) * v00 + wx * v01)
         +         wy  * ((1.0f - wx) * v10 + wx * v11);
}

// ---------------------------------------------------------------------------
// Depthwise 3x3 SAME conv + bias + ReLU on token layout [B,HW,C].
// ---------------------------------------------------------------------------
__global__ void dw_kernel(const float* __restrict__ in, const float* __restrict__ w,
                          const float* __restrict__ bias, float* __restrict__ out,
                          int B, int C, int H, int W)
{
  const long total = (long)B * H * W * C;
  long i = (long)blockIdx.x * blockDim.x + threadIdx.x;
  if (i >= total) return;
  const int c = (int)(i % C); long t = i / C;
  const int x = (int)(t % W); t /= W;
  const int y = (int)(t % H);
  const int b = (int)(t / H);
  const float* wp = w + c * 9;
  float s = bias[c];
#pragma unroll
  for (int dy = -1; dy <= 1; ++dy) {
    const int yy = y + dy;
    if (yy < 0 || yy >= H) continue;
#pragma unroll
    for (int dx = -1; dx <= 1; ++dx) {
      const int xx = x + dx;
      if (xx < 0 || xx >= W) continue;
      s += wp[(dy + 1) * 3 + (dx + 1)] * in[((long)b * H * W + yy * W + xx) * C + c];
    }
  }
  out[i] = fmaxf(s, 0.0f);
}

// ---------------------------------------------------------------------------
// Residual add + LayerNorm: out = LN(y + t) * g + b, one wave32 per token.
// ---------------------------------------------------------------------------
__global__ void ln_kernel(const float* __restrict__ y, const float* __restrict__ t,
                          const float* __restrict__ g, const float* __restrict__ b,
                          float* __restrict__ o, int Mtok, int C)
{
  const int tok  = blockIdx.x * 8 + (threadIdx.x >> 5);
  const int lane = threadIdx.x & 31;
  if (tok >= Mtok) return;
  const float* yp = y + (long)tok * C;
  const float* tp = t + (long)tok * C;
  float*       op = o + (long)tok * C;
  float s = 0.0f, s2 = 0.0f;
  for (int i = lane; i < C; i += 32) {
    const float v = yp[i] + tp[i];
    s += v; s2 += v * v;
  }
#pragma unroll
  for (int off = 16; off; off >>= 1) {
    s  += __shfl_xor(s,  off, 32);
    s2 += __shfl_xor(s2, off, 32);
  }
  const float mu  = s / C;
  const float var = s2 / C - mu * mu;
  const float rst = rsqrtf(var + 1e-5f);
  for (int i = lane; i < C; i += 32) {
    const float v = yp[i] + tp[i];
    op[i] = (v - mu) * rst * g[i] + b[i];
  }
}

// ---------------------------------------------------------------------------
// Linear-attention context: ctx[b,h,d,e] = softmax_d( scale * sum_n k*v ).
// kv: [B, N, 2C] with k at col h*64+c, v at col C+h*64+c.  One block per (b,h).
// ---------------------------------------------------------------------------
__global__ __launch_bounds__(256) void ctx_kernel(
    const float* __restrict__ kv, int Ntok, int C, int heads,
    float scale, float* __restrict__ ctx)
{
  const int z = blockIdx.x;
  const int b = z / heads;
  const int h = z - b * heads;
  const float* kbase = kv + (long)b * Ntok * (2 * C) + h * 64;
  const float* vbase = kbase + C;
  __shared__ float sm[4224];
  float* ks = sm;
  float* vs = sm + 2112;
  const int tid = threadIdx.x;
  const int dI  = tid >> 2;
  const int e0  = (tid & 3) * 16;
  float acc[16];
#pragma unroll
  for (int j = 0; j < 16; ++j) acc[j] = 0.0f;

  const int row = tid >> 3;
  const int cb  = (tid & 7) * 8;
  for (int n0 = 0; n0 < Ntok; n0 += 32) {
    __syncthreads();
    const float* kp = kbase + (long)(n0 + row) * (2 * C) + cb;
    const float* vp = vbase + (long)(n0 + row) * (2 * C) + cb;
#pragma unroll
    for (int j = 0; j < 8; ++j) { ks[row * 66 + cb + j] = kp[j]; vs[row * 66 + cb + j] = vp[j]; }
    __syncthreads();
    for (int nn = 0; nn < 32; ++nn) {
      const float kd = ks[nn * 66 + dI];
#pragma unroll
      for (int j = 0; j < 16; ++j) acc[j] += kd * vs[nn * 66 + e0 + j];
    }
  }
  __syncthreads();
#pragma unroll
  for (int j = 0; j < 16; ++j) sm[dI * 65 + e0 + j] = acc[j] * scale;
  __syncthreads();
  if (tid < 64) {
    const int e = tid;
    float mx = -1e30f;
    for (int d = 0; d < 64; ++d) mx = fmaxf(mx, sm[d * 65 + e]);
    float ssum = 0.0f;
    for (int d = 0; d < 64; ++d) ssum += expf(sm[d * 65 + e] - mx);
    const float inv = 1.0f / ssum;
    float* op = ctx + (long)z * 4096;
    for (int d = 0; d < 64; ++d) op[d * 64 + e] = expf(sm[d * 65 + e] - mx) * inv;
  }
}

// ctx [B*heads,64,64] -> dense block-diagonal [B, C, C]
__global__ void expand_kernel(const float* __restrict__ ctx, float* __restrict__ out,
                              int B, int heads, int C)
{
  const long total = (long)B * C * C;
  long i = (long)blockIdx.x * blockDim.x + threadIdx.x;
  if (i >= total) return;
  const int j = (int)(i % C); long t = i / C;
  const int r = (int)(t % C);
  const int b = (int)(t / C);
  const int hr = r >> 6, hc = j >> 6;
  out[i] = (hr == hc)
      ? ctx[((long)b * heads + hr) * 4096 + (r & 63) * 64 + (j & 63)]
      : 0.0f;
}

__global__ void concat_kernel(const float* __restrict__ a, const float* __restrict__ b,
                              float* __restrict__ out, long M, int C)
{
  const long total = M * 2 * C;
  long i = (long)blockIdx.x * blockDim.x + threadIdx.x;
  if (i >= total) return;
  const long m = i / (2 * C);
  const int  j = (int)(i - m * 2 * C);
  out[i] = (j < C) ? a[m * C + j] : b[m * C + (j - C)];
}

// tokens [B,HW,C] -> NCHW [B,C,HW]
__global__ void t2c_kernel(const float* __restrict__ t, float* __restrict__ out,
                           int B, int C, int HW)
{
  const long total = (long)B * C * HW;
  long i = (long)blockIdx.x * blockDim.x + threadIdx.x;
  if (i >= total) return;
  const int pix = (int)(i % HW); long u = i / HW;
  const int c = (int)(u % C);
  const int b = (int)(u / C);
  out[i] = t[((long)b * HW + pix) * C + c];
}

__global__ void copy_kernel(const float* __restrict__ src, float* __restrict__ dst, long n)
{
  long i = (long)blockIdx.x * blockDim.x + threadIdx.x;
  if (i < n) dst[i] = src[i];
}

// ---------------------------------------------------------------------------
// Host orchestration
// ---------------------------------------------------------------------------
extern "C" void kernel_launch(void* const* d_in, const int* in_sizes, int n_in,
                              void* d_out, int out_size, void* d_ws, size_t ws_size,
                              hipStream_t stream)
{
  (void)in_sizes; (void)n_in; (void)out_size; (void)ws_size;
  auto P = [&](int i) -> const float* { return (const float*)d_in[i]; };
  float* OUT = (float*)d_out;
  float* WS  = (float*)d_ws;
  const int  Bn = 16;
  const long TOKMAX = 16777216;           // 16*16384*64 (largest token tensor)

  float* sA = WS;                          // 4,194,304  (pre-conv low-res tmp)
  float* sB = WS + 4194304;                // TOKMAX
  float* sC = sB + TOKMAX;                 // TOKMAX
  float* sD = sC + TOKMAX;                 // 2*TOKMAX (kv1 / y1,y2)
  float* sE = sD + 2 * TOKMAX;             // 2*TOKMAX (kv2 / ffn tmps / concat)
  float* sF = sE + 2 * TOKMAX;             // ~4M      (ctx + block-diag ctx)

  auto gemm = [&](const float* A, long lda, long sAz, int chHW,
                  const float* Bm, int ldb, long sBz, int bW,
                  const float* bias, float* Cp, long ldc, long sCz,
                  int M, int N, int K, int act, int batch) {
    dim3 grid((unsigned)(M / 128), (unsigned)(N / 64), (unsigned)batch);
    gemm_wmma_kernel<<<grid, 256, 0, stream>>>(A, lda, sAz, chHW, Bm, ldb, sBz, bW,
                                               bias, Cp, ldc, sCz, M, N, K, act);
  };
  auto conv = [&](const float* In, int pb, float* Out, int Cin, int Cout, int H) {
    const int M = Bn * H * H;
    dim3 grid((unsigned)(M / 128), (unsigned)(Cout / 64), 1);
    conv3x3_wmma_kernel<<<grid, 256, 0, stream>>>(In, P(pb), P(pb + 1), P(pb + 2),
                                                  P(pb + 3), P(pb + 4), Out, Cin, Cout, H, H);
  };
  auto ew = [&](long n) { return dim3((unsigned)((n + 255) / 256)); };

  // output 0: x1 passthrough
  { const long n = 2097152; copy_kernel<<<ew(n), 256, 0, stream>>>(P(0), OUT, n); }

  // per-scale configuration (param indices follow _make_params insertion order)
  const int C_[3]   = {320, 128, 64};
  const int H_[3]   = {32, 64, 128};
  const int hd_[3]  = {5, 2, 1};
  const int CinP[3] = {512, 320, 128};
  const int Hin[3]  = {16, 32, 64};
  const int pcA[3]  = {4, 14, 24},  pcB[3] = {9, 19, 29};
  const int cpS[3]  = {40, 42, 44}, cpO[3] = {34, 36, 38};
  const int kv1[3]  = {46, 48, 50}, kv2[3] = {47, 49, 51};
  const int fS[3]   = {52, 64, 76}, fO[3]  = {58, 70, 82};
  const int lnS[3]  = {88, 92, 96}, lnO[3] = {90, 94, 98};
  const int fu[3]   = {100, 106, 112};
  const float* other[3] = {P(1), P(2), P(3)};
  const float* preIn[3] = {P(0), OUT + 2097152, OUT + 7340032};
  float* dst[3]         = {OUT + 2097152, OUT + 7340032, OUT + 15728640};

  for (int s = 0; s < 3; ++s) {
    const int C = C_[s], H = H_[s], heads = hd_[s];
    const int HW = H * H, Np = HW, M = Bn * HW;

    // ---- pre_conv: cba -> up2 -> cba ----
    conv(preIn[s], pcA[s], sA, CinP[s], C, Hin[s]);
    up2_kernel<<<ew((long)Bn * C * 4 * Hin[s] * Hin[s]), 256, 0, stream>>>(
        sA, sB, Bn, C, Hin[s], Hin[s]);
    conv(sB, pcB[s], sC, C, C, H);

    // ---- token projections (fla fused into GEMM via channel-major A) ----
    gemm(sC, 0, 0, HW, P(cpS[s]), C, 0, 1, P(cpS[s] + 1), sB, C, 0, M, C, C, 1, 1);       // r
    gemm(other[s], 0, 0, HW, P(cpO[s]), C, 0, 1, P(cpO[s] + 1), sC, C, 0, M, C, C, 1, 1); // u

    // ---- kv projections ----
    gemm(sB, C, 0, 0, P(kv1[s]), C, 0, 1, nullptr, sD, 2 * C, 0, M, 2 * C, C, 0, 1);
    gemm(sC, C, 0, 0, P(kv2[s]), C, 0, 1, nullptr, sE, 2 * C, 0, M, 2 * C, C, 0, 1);

    // ---- contexts + block-diagonal expansion ----
    const long ctxSz = (long)Bn * heads * 4096;
    float* ctx1 = sF;  float* ctx2 = sF + ctxSz;
    float* cd1  = ctx2 + ctxSz;  float* cd2 = cd1 + (long)Bn * C * C;
    ctx_kernel<<<Bn * heads, 256, 0, stream>>>(sD, Np, C, heads, 0.125f, ctx1);
    ctx_kernel<<<Bn * heads, 256, 0, stream>>>(sE, Np, C, heads, 0.125f, ctx2);
    { const long n = (long)Bn * C * C;
      expand_kernel<<<ew(n), 256, 0, stream>>>(ctx1, cd1, Bn, heads, C);
      expand_kernel<<<ew(n), 256, 0, stream>>>(ctx2, cd2, Bn, heads, C); }

    // ---- y1 = q_self @ ctx2 ; y2 = q_other @ ctx1 (batched over b) ----
    float* y1 = sD; float* y2 = sD + (long)M * C;
    gemm(sB, C, (long)Np * C, 0, cd2, C, (long)C * C, 0, nullptr,
         y1, C, (long)Np * C, Np, C, C, 0, Bn);
    gemm(sC, C, (long)Np * C, 0, cd1, C, (long)C * C, 0, nullptr,
         y2, C, (long)Np * C, Np, C, C, 0, Bn);

    // ---- FFN + residual LN, both branches ----
    float* e0 = sE; float* e1 = sE + (long)M * C;
    gemm(y1, C, 0, 0, P(fS[s]), C, 0, 1, P(fS[s] + 1), e0, C, 0, M, C, C, 0, 1);
    dw_kernel<<<ew((long)M * C), 256, 0, stream>>>(e0, P(fS[s] + 2), P(fS[s] + 3), e1, Bn, C, H, H);
    gemm(e1, C, 0, 0, P(fS[s] + 4), C, 0, 1, P(fS[s] + 5), sB, C, 0, M, C, C, 0, 1);
    ln_kernel<<<M / 8, 256, 0, stream>>>(y1, sB, P(lnS[s]), P(lnS[s] + 1), y1, M, C);

    gemm(y2, C, 0, 0, P(fO[s]), C, 0, 1, P(fO[s] + 1), e0, C, 0, M, C, C, 0, 1);
    dw_kernel<<<ew((long)M * C), 256, 0, stream>>>(e0, P(fO[s] + 2), P(fO[s] + 3), e1, Bn, C, H, H);
    gemm(e1, C, 0, 0, P(fO[s] + 4), C, 0, 1, P(fO[s] + 5), sB, C, 0, M, C, C, 0, 1);
    ln_kernel<<<M / 8, 256, 0, stream>>>(y2, sB, P(lnO[s]), P(lnO[s] + 1), y2, M, C);

    // ---- fuse: concat -> lin -> dwconv -> lin -> NCHW store ----
    concat_kernel<<<ew((long)M * 2 * C), 256, 0, stream>>>(y1, y2, sE, (long)M, C);
    gemm(sE, 2 * C, 0, 0, P(fu[s]), 2 * C, 0, 1, P(fu[s] + 1), sB, C, 0, M, C, 2 * C, 0, 1);
    dw_kernel<<<ew((long)M * C), 256, 0, stream>>>(sB, P(fu[s] + 2), P(fu[s] + 3), sC, Bn, C, H, H);
    gemm(sC, C, 0, 0, P(fu[s] + 4), C, 0, 1, P(fu[s] + 5), sB, C, 0, M, C, C, 0, 1);
    t2c_kernel<<<ew((long)M * C), 256, 0, stream>>>(sB, dst[s], Bn, C, HW);
  }
}
